// GraphBiasedAttention_12094627905800
// MI455X (gfx1250) — matrix operand order repaired and checked by reading, not compiled, not run on hardware
//
#include <hip/hip_runtime.h>
#include <hip/hip_bf16.h>
#include <math.h>

typedef __attribute__((ext_vector_type(16))) _Float16 v16h;
typedef __attribute__((ext_vector_type(8)))  _Float16 v8h;
typedef __attribute__((ext_vector_type(8)))  float    v8f;

#define HIDDEN 1024
#define NH 16
#define HD 64
#define SEQ 2048
#define BATCH 2

// A operand (16x32 f16): lane = m + 16*hh. halves 0-7 = K[hh*16..hh*16+7],
// halves 8-15 = K[hh*16+8..hh*16+15]  (contiguous k = 16*hh + s, same as make_b)
__device__ __forceinline__ v16h make_a(const _Float16* p0, const _Float16* p1) {
    v8h lo = *(const v8h*)p0;
    v8h hi = *(const v8h*)p1;
    v16h r;
#pragma unroll
    for (int i = 0; i < 8; ++i) { r[i] = lo[i]; r[8 + i] = hi[i]; }
    return r;
}

// B operand (32x16 f16): lane = n + 16*g. halves t = B[K=16g+t][n].
// For B = (row-major M)^T this is one contiguous 32B chunk of row n at K offset 16g.
__device__ __forceinline__ v16h make_b(const _Float16* p) {
    v8h lo = *(const v8h*)p;
    v8h hi = *(const v8h*)(p + 8);
    v16h r;
#pragma unroll
    for (int i = 0; i < 8; ++i) { r[i] = lo[i]; r[8 + i] = hi[i]; }
    return r;
}

__global__ void cvt_f32_f16(const float* __restrict__ src, _Float16* __restrict__ dst, int n) {
    for (int i = blockIdx.x * blockDim.x + threadIdx.x; i < n; i += gridDim.x * blockDim.x)
        dst[i] = (_Float16)src[i];
}

// out = X(4096x1024,f16) @ W^T + bias, scattered to per-head f16 layout.
// transposeV==0 : out[((b*16+h)*2048 + s)*64 + d]   (Q/K layout)
// transposeV==1 : out[((b*16+h)*64 + d)*2048 + s]   (V transposed for PV B-operand)
__global__ void gemm_qkv(const _Float16* __restrict__ X, const _Float16* __restrict__ W,
                         const float* __restrict__ bias, _Float16* __restrict__ out,
                         int transposeV) {
    int wave = (blockIdx.x * blockDim.x + threadIdx.x) >> 5;  // one 16x64 tile per wave
    int lane = threadIdx.x & 31;
    int m16 = lane & 15, hh = lane >> 4, g16 = hh * 16;
    int tm = wave >> 4;   // 256 M-tiles
    int tn = wave & 15;   // 16 N-tiles of 64 (one head each)

    const _Float16* xrow  = X + (size_t)(tm * 16 + m16) * HIDDEN;
    const _Float16* wrow0 = W + (size_t)(tn * 64 + m16) * HIDDEN;
    v8f acc[4] = {};

    v16h a  = make_a(xrow + g16, xrow + 8 + g16);
    v16h b0 = make_b(wrow0 + 0 * 16 * HIDDEN + g16);
    v16h b1 = make_b(wrow0 + 1 * 16 * HIDDEN + g16);
    v16h b2 = make_b(wrow0 + 2 * 16 * HIDDEN + g16);
    v16h b3 = make_b(wrow0 + 3 * 16 * HIDDEN + g16);
    for (int kk = 32; kk < HIDDEN; kk += 32) {
        v16h an  = make_a(xrow + kk + g16, xrow + kk + 8 + g16);
        v16h bn0 = make_b(wrow0 + 0 * 16 * HIDDEN + kk + g16);
        v16h bn1 = make_b(wrow0 + 1 * 16 * HIDDEN + kk + g16);
        v16h bn2 = make_b(wrow0 + 2 * 16 * HIDDEN + kk + g16);
        v16h bn3 = make_b(wrow0 + 3 * 16 * HIDDEN + kk + g16);
        acc[0] = __builtin_amdgcn_wmma_f32_16x16x32_f16(false, a, false, b0, (short)0, acc[0], false, false);
        acc[1] = __builtin_amdgcn_wmma_f32_16x16x32_f16(false, a, false, b1, (short)0, acc[1], false, false);
        acc[2] = __builtin_amdgcn_wmma_f32_16x16x32_f16(false, a, false, b2, (short)0, acc[2], false, false);
        acc[3] = __builtin_amdgcn_wmma_f32_16x16x32_f16(false, a, false, b3, (short)0, acc[3], false, false);
        a = an; b0 = bn0; b1 = bn1; b2 = bn2; b3 = bn3;
    }
    acc[0] = __builtin_amdgcn_wmma_f32_16x16x32_f16(false, a, false, b0, (short)0, acc[0], false, false);
    acc[1] = __builtin_amdgcn_wmma_f32_16x16x32_f16(false, a, false, b1, (short)0, acc[1], false, false);
    acc[2] = __builtin_amdgcn_wmma_f32_16x16x32_f16(false, a, false, b2, (short)0, acc[2], false, false);
    acc[3] = __builtin_amdgcn_wmma_f32_16x16x32_f16(false, a, false, b3, (short)0, acc[3], false, false);

#pragma unroll
    for (int t = 0; t < 4; ++t) {
        int n = tn * 64 + t * 16 + m16;
        float bv = bias[n];
        int h = n >> 6, d = n & 63;
#pragma unroll
        for (int r = 0; r < 8; ++r) {
            int m = tm * 16 + r + 8 * hh;
            int b = m >> 11, s = m & 2047;
            float v = acc[t][r] + bv;
            size_t off = transposeV ? (((size_t)(b * NH + h) * HD + d) * SEQ + s)
                                    : (((size_t)(b * NH + h) * SEQ + s) * HD + d);
            out[off] = (_Float16)v;
        }
    }
}

// out(f32, [4096,1024]) = OH(f16) @ Wo^T + bo
__global__ void gemm_out(const _Float16* __restrict__ X, const _Float16* __restrict__ W,
                         const float* __restrict__ bias, float* __restrict__ out) {
    int wave = (blockIdx.x * blockDim.x + threadIdx.x) >> 5;
    int lane = threadIdx.x & 31;
    int m16 = lane & 15, hh = lane >> 4, g16 = hh * 16;
    int tm = wave >> 4, tn = wave & 15;

    const _Float16* xrow  = X + (size_t)(tm * 16 + m16) * HIDDEN;
    const _Float16* wrow0 = W + (size_t)(tn * 64 + m16) * HIDDEN;
    v8f acc[4] = {};

    v16h a  = make_a(xrow + g16, xrow + 8 + g16);
    v16h b0 = make_b(wrow0 + 0 * 16 * HIDDEN + g16);
    v16h b1 = make_b(wrow0 + 1 * 16 * HIDDEN + g16);
    v16h b2 = make_b(wrow0 + 2 * 16 * HIDDEN + g16);
    v16h b3 = make_b(wrow0 + 3 * 16 * HIDDEN + g16);
    for (int kk = 32; kk < HIDDEN; kk += 32) {
        v16h an  = make_a(xrow + kk + g16, xrow + kk + 8 + g16);
        v16h bn0 = make_b(wrow0 + 0 * 16 * HIDDEN + kk + g16);
        v16h bn1 = make_b(wrow0 + 1 * 16 * HIDDEN + kk + g16);
        v16h bn2 = make_b(wrow0 + 2 * 16 * HIDDEN + kk + g16);
        v16h bn3 = make_b(wrow0 + 3 * 16 * HIDDEN + kk + g16);
        acc[0] = __builtin_amdgcn_wmma_f32_16x16x32_f16(false, a, false, b0, (short)0, acc[0], false, false);
        acc[1] = __builtin_amdgcn_wmma_f32_16x16x32_f16(false, a, false, b1, (short)0, acc[1], false, false);
        acc[2] = __builtin_amdgcn_wmma_f32_16x16x32_f16(false, a, false, b2, (short)0, acc[2], false, false);
        acc[3] = __builtin_amdgcn_wmma_f32_16x16x32_f16(false, a, false, b3, (short)0, acc[3], false, false);
        a = an; b0 = bn0; b1 = bn1; b2 = bn2; b3 = bn3;
    }
    acc[0] = __builtin_amdgcn_wmma_f32_16x16x32_f16(false, a, false, b0, (short)0, acc[0], false, false);
    acc[1] = __builtin_amdgcn_wmma_f32_16x16x32_f16(false, a, false, b1, (short)0, acc[1], false, false);
    acc[2] = __builtin_amdgcn_wmma_f32_16x16x32_f16(false, a, false, b2, (short)0, acc[2], false, false);
    acc[3] = __builtin_amdgcn_wmma_f32_16x16x32_f16(false, a, false, b3, (short)0, acc[3], false, false);

#pragma unroll
    for (int t = 0; t < 4; ++t) {
        int n = tn * 64 + t * 16 + m16;
        float bv = bias[n];
#pragma unroll
        for (int r = 0; r < 8; ++r) {
            int m = tm * 16 + r + 8 * hh;
            out[(size_t)m * HIDDEN + n] = acc[t][r] + bv;
        }
    }
}

// One workgroup = (bh, 16-query block). 128KB f32 score stripe in CDNA5's 320KB LDS.
__global__ void attn_kernel(const _Float16* __restrict__ Qh, const _Float16* __restrict__ Kh,
                            const _Float16* __restrict__ Vt,
                            const int* __restrict__ dist, const float* __restrict__ idf,
                            const int* __restrict__ amask, const float* __restrict__ semb,
                            const float* __restrict__ rwp,
                            float* __restrict__ attn_out, _Float16* __restrict__ outh) {
    __shared__ float sc[16 * SEQ];    // 128 KB score/prob stripe
    __shared__ float se[64];          // struct_emb column for this head
    __shared__ float rmax[16], rsum[16];
    __shared__ float oacc[16 * HD];   // PV reduction tile

    int qb = blockIdx.x;              // 0..127
    int bh = blockIdx.y;              // 0..31
    int b = bh >> 4, h = bh & 15;
    int tid = threadIdx.x;
    int wave = tid >> 5, lane = tid & 31;
    int m16 = lane & 15, hh = lane >> 4, g16 = hh * 16;
    float rweight = rwp[0];

    if (tid < 51) se[tid] = semb[tid * NH + h];
    __syncthreads();

    // Q tile A-operands (K-dim 64 -> 2 WMMA k-steps), shared by all key tiles
    const _Float16* qrow = Qh + ((size_t)bh * SEQ + qb * 16 + m16) * HD;
    v16h aq0 = make_a(qrow + g16,      qrow + 8 + g16);
    v16h aq1 = make_a(qrow + 32 + g16, qrow + 40 + g16);

    float idfq[8];
#pragma unroll
    for (int r = 0; r < 8; ++r) idfq[r] = idf[b * SEQ + qb * 16 + r + 8 * hh];

    const _Float16* kbase = Kh + (size_t)bh * SEQ * HD;
    // each of 8 waves computes 16 key-tiles of 16 keys; pipeline K loads one deep
    {
        const _Float16* krow = kbase + (size_t)(wave * 256 + m16) * HD;
        v16h bk0 = make_b(krow + g16);
        v16h bk1 = make_b(krow + 32 + g16);
        for (int i = 0; i < 16; ++i) {
            int inext = (i < 15) ? i + 1 : i;
            const _Float16* krn = kbase + (size_t)(wave * 256 + inext * 16 + m16) * HD;
            v16h bn0 = make_b(krn + g16);
            v16h bn1 = make_b(krn + 32 + g16);

            v8f c = {};
            c = __builtin_amdgcn_wmma_f32_16x16x32_f16(false, aq0, false, bk0, (short)0, c, false, false);
            c = __builtin_amdgcn_wmma_f32_16x16x32_f16(false, aq1, false, bk1, (short)0, c, false, false);

            int kg0 = (wave * 16 + i) * 16;
            int kg = kg0 + m16;
            float idfk = idf[b * SEQ + kg];
            int mk = amask[b * SEQ + kg];
            const int* drow = dist + ((size_t)b * SEQ + qb * 16) * SEQ + kg;
#pragma unroll
            for (int r = 0; r < 8; ++r) {
                int ql = r + 8 * hh;
                int dv = drow[(size_t)ql * SEQ];
                dv = dv < 0 ? 0 : (dv > 50 ? 50 : dv);
                float sv = c[r] * 0.125f + se[dv] + rweight * (idfk + idfq[r]);
                if (mk == 0) sv = -INFINITY;
                sc[ql * SEQ + kg0 + m16] = sv;
            }
            bk0 = bn0; bk1 = bn1;
        }
    }
    __syncthreads();

    // softmax stats: 16 threads per row, half-wave shuffle reduction
    {
        int row = tid >> 4, t = tid & 15;
        const float* p = sc + row * SEQ;
        float mx = -INFINITY;
        for (int c2 = t; c2 < SEQ; c2 += 16) mx = fmaxf(mx, p[c2]);
#pragma unroll
        for (int o = 8; o >= 1; o >>= 1) mx = fmaxf(mx, __shfl_xor(mx, o, 16));
        float sum = 0.f;
        for (int c2 = t; c2 < SEQ; c2 += 16) sum += __expf(p[c2] - mx);
#pragma unroll
        for (int o = 8; o >= 1; o >>= 1) sum += __shfl_xor(sum, o, 16);
        if (t == 0) { rmax[row] = mx; rsum[row] = 1.0f / sum; }
    }
    __syncthreads();

    // normalize in LDS + non-temporal coalesced attn writeback (537 MB stream;
    // keep L2 for the K/V/dist working set instead)
    float* aout = attn_out + ((size_t)bh * SEQ + qb * 16) * SEQ;
    for (int idx = tid; idx < 16 * SEQ; idx += 256) {
        int row = idx >> 11;
        float pv = __expf(sc[idx] - rmax[row]) * rsum[row];
        sc[idx] = pv;
        __builtin_nontemporal_store(pv, aout + idx);
    }
    for (int idx = tid; idx < 16 * HD; idx += 256) oacc[idx] = 0.f;
    __syncthreads();

    // PV: wave w -> (d-tile = w&3, k-half = w>>2); probs from LDS (cvt to f16,
    // co-executes with XDL), V^T from global, pipelined one deep
    {
        int nt = wave & 3, kh2 = wave >> 2;
        int jbeg = kh2 * 32, jend = jbeg + 32;
        const _Float16* vrow = Vt + ((size_t)bh * HD + nt * 16 + m16) * SEQ;
        v8f c = {};
        v16h bv = make_b(vrow + jbeg * 32 + g16);
        for (int j = jbeg; j < jend; ++j) {
            int jn = (j + 1 < jend) ? j + 1 : j;
            v16h bvn = make_b(vrow + jn * 32 + g16);
            int k0 = j * 32;
            const float* prow = sc + m16 * SEQ + k0 + g16;
            v16h a;
#pragma unroll
            for (int i2 = 0; i2 < 8; ++i2) {
                a[i2] = (_Float16)prow[i2];
                a[8 + i2] = (_Float16)prow[8 + i2];
            }
            c = __builtin_amdgcn_wmma_f32_16x16x32_f16(false, a, false, bv, (short)0, c, false, false);
            bv = bvn;
        }
#pragma unroll
        for (int r = 0; r < 8; ++r)
            atomicAdd(&oacc[(r + 8 * hh) * HD + nt * 16 + m16], c[r]);
    }
    __syncthreads();

    // out_heads (f16, [B*S, 1024]) feeding the Wo GEMM
    for (int idx = tid; idx < 16 * HD; idx += 256) {
        int ql = idx >> 6, d = idx & 63;
        size_t m = (size_t)b * SEQ + qb * 16 + ql;
        outh[m * HIDDEN + h * HD + d] = (_Float16)oacc[idx];
    }
}

extern "C" void kernel_launch(void* const* d_in, const int* in_sizes, int n_in,
                              void* d_out, int out_size, void* d_ws, size_t ws_size,
                              hipStream_t stream) {
    const float* hs   = (const float*)d_in[0];
    const int*   dist = (const int*)  d_in[1];
    const float* idf  = (const float*)d_in[2];
    const int*   am   = (const int*)  d_in[3];
    const float* Wq   = (const float*)d_in[4];
    const float* bq   = (const float*)d_in[5];
    const float* Wk   = (const float*)d_in[6];
    const float* bk   = (const float*)d_in[7];
    const float* Wv   = (const float*)d_in[8];
    const float* bv   = (const float*)d_in[9];
    const float* Wo   = (const float*)d_in[10];
    const float* bo   = (const float*)d_in[11];
    const float* semb = (const float*)d_in[12];
    const float* rw   = (const float*)d_in[13];

    const int NX = BATCH * SEQ * HIDDEN;   // 4194304
    const int NW = HIDDEN * HIDDEN;        // 1048576
    _Float16* Xh  = (_Float16*)d_ws;
    _Float16* Wqh = Xh  + NX;
    _Float16* Wkh = Wqh + NW;
    _Float16* Wvh = Wkh + NW;
    _Float16* Woh = Wvh + NW;
    _Float16* Qh  = Woh + NW;
    _Float16* Kh  = Qh  + NX;
    _Float16* Vt  = Kh  + NX;
    _Float16* OH  = Vt  + NX;

    cvt_f32_f16<<<1024, 256, 0, stream>>>(hs, Xh, NX);
    cvt_f32_f16<<<512, 256, 0, stream>>>(Wq, Wqh, NW);
    cvt_f32_f16<<<512, 256, 0, stream>>>(Wk, Wkh, NW);
    cvt_f32_f16<<<512, 256, 0, stream>>>(Wv, Wvh, NW);
    cvt_f32_f16<<<512, 256, 0, stream>>>(Wo, Woh, NW);

    gemm_qkv<<<512, 256, 0, stream>>>(Xh, Wqh, bq, Qh, 0);
    gemm_qkv<<<512, 256, 0, stream>>>(Xh, Wkh, bk, Kh, 0);
    gemm_qkv<<<512, 256, 0, stream>>>(Xh, Wvh, bv, Vt, 1);

    float* outp  = (float*)d_out;
    float* attnp = outp + (size_t)NX;
    dim3 grid(SEQ / 16, BATCH * NH);
    attn_kernel<<<grid, 256, 0, stream>>>(Qh, Kh, Vt, dist, idf, am, semb, rw, attnp, OH);

    gemm_out<<<512, 256, 0, stream>>>(OH, Woh, bo, outp);
}
